// AttentionLayer_57320633532721
// MI455X (gfx1250) — matrix operand order repaired and checked by reading, not compiled, not run on hardware
//
#include <hip/hip_runtime.h>

#define DIM   512
#define SEQ   2048
#define BATCH 8
#define BT    128     // block tile (M and N)
#define KT    32      // k-chunk per WMMA
#define LDSS  40      // padded LDS row stride in bf16 elements

typedef __bf16 bf16;
typedef __attribute__((ext_vector_type(16))) __bf16 v16bf;
typedef __attribute__((ext_vector_type(8)))  __bf16 v8bf;
typedef __attribute__((ext_vector_type(4)))  __bf16 v4bf;
typedef __attribute__((ext_vector_type(8)))  float  v8f;

union FragAB { v16bf v; v8bf h[2]; };

// ---------------- gfx1250 async Global->LDS path (guarded) -----------------
#if __has_builtin(__builtin_amdgcn_global_load_async_to_lds_b128)
#define HAVE_ASYNC_LDS 1
#else
#define HAVE_ASYNC_LDS 0
#endif

// The builtin expects pointers to 16-byte int vectors (AS1 src, AS3 dst).
typedef int b128v __attribute__((vector_size(4 * sizeof(int))));
#define GPTR(p) ((__attribute__((address_space(1))) b128v*)(p))
#define LPTR(p) ((__attribute__((address_space(3))) b128v*)(p))

__device__ __forceinline__ void wait_async_lds() {
#if HAVE_ASYNC_LDS
#if __has_builtin(__builtin_amdgcn_s_wait_asynccnt)
    __builtin_amdgcn_s_wait_asynccnt(0);
#else
    asm volatile("s_wait_asynccnt 0x0" ::: "memory");
#endif
#endif
}

// Copy a 128x32 bf16 tile (row-major, leading dim ld) into LDS (stride LDSS).
// Async (ASYNCcnt) when available; caller must wait_async_lds() + barrier.
__device__ __forceinline__ void copy_tile_rows_async(bf16* lds, const bf16* src, int ld, int t) {
#pragma unroll
    for (int it = 0; it < 2; ++it) {
        int r = it * 64 + (t >> 2);
        int q = (t & 3) * 8;
#if HAVE_ASYNC_LDS
        __builtin_amdgcn_global_load_async_to_lds_b128(
            GPTR(src + (size_t)r * ld + q),
            LPTR(&lds[r * LDSS + q]),
            0, 0);
#else
        *(v8bf*)(&lds[r * LDSS + q]) = *(const v8bf*)(src + (size_t)r * ld + q);
#endif
    }
}

// ---- WMMA fragment loads from LDS (wave32 layouts per CDNA5 ISA 7.12.2) ----

// A: 16x32 bf16. lane l (0-15): row M=l, K=0..7 in h[0], K=16..23 in h[1].
//    lane l+16: row M=l, K=8..15 in h[0], K=24..31 in h[1].
__device__ __forceinline__ v16bf lds_frag_A(const bf16* A, int row0, int lane) {
    int r = row0 + (lane & 15);
    int half = lane >> 4;
    const bf16* p = A + r * LDSS + half * 8;
    FragAB f;
    f.h[0] = *(const v8bf*)(p);
    f.h[1] = *(const v8bf*)(p + 16);
    return f.v;
}

// B: 32x16 bf16 stored column-major in LDS ([n][k], stride LDSS).
//    lane l (0-15): col N=l, K=0..7 in h[0], K=16..23 in h[1]; lane l+16: col N=l, K=8..15 in h[0], K=24..31 in h[1].
__device__ __forceinline__ v16bf lds_frag_B(const bf16* B, int col0, int lane) {
    int n = col0 + (lane & 15);
    const bf16* p = B + n * LDSS + (lane >> 4) * 8;
    FragAB f;
    f.h[0] = *(const v8bf*)(p);
    f.h[1] = *(const v8bf*)(p + 16);
    return f.v;
}

// 8-WMMA step: wave computes its 32x64 slab from LDS tiles.
__device__ __forceinline__ void wmma_step(const bf16* ldsA, const bf16* ldsB,
                                          int wm, int wn, int lane, v8f acc[2][4]) {
    v16bf a[2], bfr[4];
    a[0] = lds_frag_A(ldsA, wm * 32, lane);
    a[1] = lds_frag_A(ldsA, wm * 32 + 16, lane);
#pragma unroll
    for (int nt = 0; nt < 4; ++nt) bfr[nt] = lds_frag_B(ldsB, wn * 64 + nt * 16, lane);
#pragma unroll
    for (int mt = 0; mt < 2; ++mt)
#pragma unroll
        for (int nt = 0; nt < 4; ++nt)
            acc[mt][nt] = __builtin_amdgcn_wmma_f32_16x16x32_bf16(
                false, a[mt], false, bfr[nt], (short)0, acc[mt][nt], false, false);
}

__device__ __forceinline__ void zero_acc(v8f acc[2][4]) {
#pragma unroll
    for (int i = 0; i < 2; ++i)
#pragma unroll
        for (int j = 0; j < 4; ++j)
#pragma unroll
            for (int r = 0; r < 8; ++r) acc[i][j][r] = 0.0f;
}

// ---------------------------------------------------------------------------
// Kernel 1: QKV projection.  C[m0:+128, n0:+128] = x . W  (bf16 out).
// z==0 -> Q (scaled by 1/sqrt(d)), z==1 -> K, z==2 -> V stored TRANSPOSED
// per batch as Vt[b][d][seq].
// grid = (DIM/BT, M/BT, 3)  block = 256
// ---------------------------------------------------------------------------
__global__ __launch_bounds__(256) void qkv_kernel(
    const float* __restrict__ x,
    const float* __restrict__ Wq, const float* __restrict__ Wk, const float* __restrict__ Wv,
    bf16* __restrict__ Q, bf16* __restrict__ Kd, bf16* __restrict__ Vt)
{
    __shared__ bf16 ldsA[BT * LDSS];
    __shared__ bf16 ldsB[BT * LDSS];
    const int t = threadIdx.x;
    const int lane = t & 31, w = t >> 5;
    const int wm = w & 3, wn = w >> 2;
    const int m0 = blockIdx.y * BT;
    const int n0 = blockIdx.x * BT;

    const float* W;
    if (blockIdx.z == 0)      W = Wq;
    else if (blockIdx.z == 1) W = Wk;
    else                      W = Wv;

    v8f acc[2][4];
    zero_acc(acc);

    for (int kc = 0; kc < DIM; kc += KT) {
        // A tile: x[m0:+128, kc:+32] fp32 -> bf16 into LDS (row-major)
#pragma unroll
        for (int it = 0; it < 4; ++it) {
            int row = it * 32 + (t >> 3);
            int col = (t & 7) * 4;
            float4 val = *(const float4*)(x + (size_t)(m0 + row) * DIM + kc + col);
            v4bf b;
            b.x = (__bf16)val.x; b.y = (__bf16)val.y; b.z = (__bf16)val.z; b.w = (__bf16)val.w;
            *(v4bf*)(&ldsA[row * LDSS + col]) = b;
        }
        // B tile: W[kc:+32, n0:+128] fp32 -> bf16, stored transposed ([n][k])
#pragma unroll
        for (int it = 0; it < 4; ++it) {
            int k = it * 8 + (t >> 5);
            int n = (t & 31) * 4;
            float4 val = *(const float4*)(W + (size_t)(kc + k) * DIM + n0 + n);
            ldsB[(n + 0) * LDSS + k] = (__bf16)val.x;
            ldsB[(n + 1) * LDSS + k] = (__bf16)val.y;
            ldsB[(n + 2) * LDSS + k] = (__bf16)val.z;
            ldsB[(n + 3) * LDSS + k] = (__bf16)val.w;
        }
        __syncthreads();
        wmma_step(ldsA, ldsB, wm, wn, lane, acc);
        __syncthreads();
    }

    // ---- Writeback (branch on uniform blockIdx.z hoisted OUT of the loops).
    // C f32 layout: M = r + 8*(lane>=16), N = lane&15.
    const int rbase0 = m0 + wm * 32 + ((lane >> 4) << 3);
    const int col0   = n0 + wn * 64 + (lane & 15);

    if (blockIdx.z < 2) {
        bf16* O = (blockIdx.z == 0) ? Q : Kd;
        const float scale = (blockIdx.z == 0) ? 0.04419417382415922f : 1.0f; // 512^-0.5
#pragma unroll
        for (int mt = 0; mt < 2; ++mt) {
            bf16* orow = O + (size_t)(rbase0 + mt * 16) * DIM + col0;
#pragma unroll
            for (int nt = 0; nt < 4; ++nt)
#pragma unroll
                for (int r = 0; r < 8; ++r)
                    orow[(size_t)r * DIM + nt * 16] = (__bf16)(acc[mt][nt][r] * scale);
        }
    } else {
        // V transposed: Vt[b][d][seq]; whole 128-row tile lies in one batch.
        const int bb = m0 >> 11;                 // batch index
        const int sbase = (rbase0 & (SEQ - 1));  // sequence position of r=0
        bf16* vb = Vt + (size_t)bb * DIM * SEQ;
#pragma unroll
        for (int mt = 0; mt < 2; ++mt) {
            bf16* vcol = vb + (size_t)col0 * SEQ + sbase + mt * 16;
#pragma unroll
            for (int nt = 0; nt < 4; ++nt)
#pragma unroll
                for (int r = 0; r < 8; ++r)
                    vcol[(size_t)(nt * 16) * SEQ + r] = (__bf16)acc[mt][nt][r];
        }
    }
}

// ---------------------------------------------------------------------------
// Generic double-buffered bf16 GEMM core: both A and B tiles are straight
// row copies (async Global->LDS when available).
//   C[m][n] = sum_k Asrc[m][k] * Bsrc[n][k]
// ---------------------------------------------------------------------------
template <int KDIM>
__device__ __forceinline__ void gemm_copy_core(
    const bf16* __restrict__ Asrc, int lda,
    const bf16* __restrict__ Bsrc, int ldb,
    float* __restrict__ C, int ldc,
    int m0, int n0)
{
    __shared__ bf16 ldsA[2][BT * LDSS];
    __shared__ bf16 ldsB[2][BT * LDSS];
    const int t = threadIdx.x;
    const int lane = t & 31, w = t >> 5;
    const int wm = w & 3, wn = w >> 2;

    v8f acc[2][4];
    zero_acc(acc);

    // prologue: fetch chunk 0
    copy_tile_rows_async(ldsA[0], Asrc + (size_t)m0 * lda, lda, t);
    copy_tile_rows_async(ldsB[0], Bsrc + (size_t)n0 * ldb, ldb, t);
    wait_async_lds();
    __syncthreads();

    for (int ki = 0; ki < KDIM / KT; ++ki) {
        const int cur = ki & 1, nxt = cur ^ 1;
        const int kn = (ki + 1) * KT;
        if (kn < KDIM) {
            copy_tile_rows_async(ldsA[nxt], Asrc + (size_t)m0 * lda + kn, lda, t);
            copy_tile_rows_async(ldsB[nxt], Bsrc + (size_t)n0 * ldb + kn, ldb, t);
        }
        wmma_step(ldsA[cur], ldsB[cur], wm, wn, lane, acc);
        wait_async_lds();
        __syncthreads();
    }

    const int rbase0 = m0 + wm * 32 + ((lane >> 4) << 3);
    const int col0   = n0 + wn * 64 + (lane & 15);
#pragma unroll
    for (int mt = 0; mt < 2; ++mt) {
        float* crow = C + (size_t)(rbase0 + mt * 16) * ldc + col0;
#pragma unroll
        for (int nt = 0; nt < 4; ++nt)
#pragma unroll
            for (int r = 0; r < 8; ++r)
                crow[(size_t)r * ldc + nt * 16] = acc[mt][nt][r];
    }
}

// Kernel 2: scores  S = Q_b . K_b^T  (Q pre-scaled), f32 out.
// grid = (SEQ/BT, SEQ/BT)  block = 256
__global__ __launch_bounds__(256) void scores_kernel(
    const bf16* __restrict__ Q, const bf16* __restrict__ Km, float* __restrict__ S)
{
    gemm_copy_core<DIM>(Q, DIM, Km, DIM, S, SEQ,
                        blockIdx.y * BT, blockIdx.x * BT);
}

// Kernel 4: O = P . V_b  ==  C[m][n] = sum_k P[m][k] * Vt[n][k], f32 out.
// grid = (DIM/BT, SEQ/BT)  block = 256
__global__ __launch_bounds__(256) void out_kernel(
    const bf16* __restrict__ P, const bf16* __restrict__ Vt, float* __restrict__ O)
{
    gemm_copy_core<SEQ>(P, SEQ, Vt, SEQ, O, DIM,
                        blockIdx.y * BT, blockIdx.x * BT);
}

// ---------------------------------------------------------------------------
// Kernel 3: row softmax, S f32 [SEQ,SEQ] -> P bf16 [SEQ,SEQ]
// grid = SEQ  block = 256
// ---------------------------------------------------------------------------
__global__ __launch_bounds__(256) void softmax_kernel(
    const float* __restrict__ S, bf16* __restrict__ P)
{
    __shared__ float red[256];
    const int row = blockIdx.x;
    const int t = threadIdx.x;
    const float* s = S + (size_t)row * SEQ;

    float vals[8];
    float m = -3.402823466e+38f;
#pragma unroll
    for (int i = 0; i < 8; ++i) {
        vals[i] = s[t + i * 256];
        m = fmaxf(m, vals[i]);
    }
    red[t] = m;
    __syncthreads();
    for (int off = 128; off > 0; off >>= 1) {
        if (t < off) red[t] = fmaxf(red[t], red[t + off]);
        __syncthreads();
    }
    m = red[0];
    __syncthreads();

    float sum = 0.0f;
#pragma unroll
    for (int i = 0; i < 8; ++i) {
        vals[i] = __expf(vals[i] - m);
        sum += vals[i];
    }
    red[t] = sum;
    __syncthreads();
    for (int off = 128; off > 0; off >>= 1) {
        if (t < off) red[t] += red[t + off];
        __syncthreads();
    }
    float inv = 1.0f / red[0];

    bf16* p = P + (size_t)row * SEQ;
#pragma unroll
    for (int i = 0; i < 8; ++i)
        p[t + i * 256] = (__bf16)(vals[i] * inv);
}

// ---------------------------------------------------------------------------
extern "C" void kernel_launch(void* const* d_in, const int* in_sizes, int n_in,
                              void* d_out, int out_size, void* d_ws, size_t ws_size,
                              hipStream_t stream)
{
    const float* x  = (const float*)d_in[0];
    const float* Wq = (const float*)d_in[1];
    const float* Wk = (const float*)d_in[2];
    const float* Wv = (const float*)d_in[3];
    float* out = (float*)d_out;

    char* ws = (char*)d_ws;
    const size_t MB = 1024 * 1024;
    bf16*  Q  = (bf16*)(ws);                // 16 MB  [b][seq][d]  (pre-scaled by 1/sqrt(d))
    bf16*  K  = (bf16*)(ws + 16 * MB);      // 16 MB  [b][seq][d]
    bf16*  Vt = (bf16*)(ws + 32 * MB);      // 16 MB  [b][d][seq]  (transposed V)
    float* S  = (float*)(ws + 48 * MB);     // 16 MB  [seq][seq] f32, reused per batch
    bf16*  P  = (bf16*)(ws + 64 * MB);      //  8 MB  [seq][seq] bf16, reused per batch

    // QKV projections for all batches at once (M = 8*2048 = 16384)
    qkv_kernel<<<dim3(DIM / BT, (BATCH * SEQ) / BT, 3), 256, 0, stream>>>(
        x, Wq, Wk, Wv, Q, K, Vt);

    for (int b = 0; b < BATCH; ++b) {
        const bf16* Qb  = Q  + (size_t)b * SEQ * DIM;
        const bf16* Kb  = K  + (size_t)b * SEQ * DIM;
        const bf16* Vtb = Vt + (size_t)b * SEQ * DIM;
        scores_kernel<<<dim3(SEQ / BT, SEQ / BT), 256, 0, stream>>>(Qb, Kb, S);
        softmax_kernel<<<SEQ, 256, 0, stream>>>(S, P);
        out_kernel<<<dim3(DIM / BT, SEQ / BT), 256, 0, stream>>>(P, Vtb, out + (size_t)b * SEQ * DIM);
    }
}
